// FullAttention_26190710571412
// MI455X (gfx1250) — compile-verified, matrix operand order checked by reading, not run on hardware
//
#include <hip/hip_runtime.h>

// ---------------------------------------------------------------------------
// Types
// ---------------------------------------------------------------------------
typedef __bf16 bf16;
typedef __attribute__((ext_vector_type(16))) __bf16       v16bf;
typedef __attribute__((ext_vector_type(8)))  float        v8f;
typedef __attribute__((ext_vector_type(4)))  unsigned int v4u;

static __device__ __forceinline__ bf16 f2bf(float f) {
  unsigned u = __builtin_bit_cast(unsigned, f);
  u += 0x7FFFu + ((u >> 16) & 1u);           // round-to-nearest-even
  unsigned short s = (unsigned short)(u >> 16);
  return __builtin_bit_cast(bf16, s);
}

// LDS byte-offset of a __shared__ object (generic -> AS3 -> int)
static __device__ __forceinline__ unsigned lds_addr_of(const void* p) {
  return (unsigned)(unsigned long long)(const __attribute__((address_space(3))) void*)p;
}

// One 16-byte async DMA: global -> LDS (tracked by ASYNCcnt, no VGPRs)
static __device__ __forceinline__ void async_b128(unsigned lds_off, const bf16* g) {
  asm volatile("global_load_async_to_lds_b128 %0, %1, off"
               :: "v"(lds_off), "v"((unsigned long long)g) : "memory");
}

// ---------------------------------------------------------------------------
// f32 -> bf16 conversion kernel
// ---------------------------------------------------------------------------
__global__ void cvt_f32_bf16(const float* __restrict__ in, bf16* __restrict__ out, long n) {
  long i = (long)blockIdx.x * blockDim.x + threadIdx.x;
  long stride = (long)gridDim.x * blockDim.x;
  for (; i < n; i += stride) out[i] = f2bf(in[i]);
}

// ---------------------------------------------------------------------------
// Common GEMM argument block:  out = alpha * (A[MxK] @ W[NxK]^T) + bias[N]
// Batch: blockIdx.z = batch*ksplit + sk;  batch decomposed zhi=batch/zdiv,
// zlo=batch%zdiv with per-pointer strides (elements).  ksplit>1 partitions K
// into chunks (bias must be null in that mode); each split writes an f32
// partial at outF + sk*sFsp (reduced later in fixed order -> deterministic).
// ---------------------------------------------------------------------------
struct GemmArgs {
  const bf16* A; long sAhi, sAlo; int lda;
  const bf16* W; long sWhi, sWlo; int ldw;
  const float* bias;
  float* outF; long sFhi, sFlo; int ldf;
  bf16*  outH; long sHhi, sHlo; int ldh;
  int transH;
  int K; float alpha; int zdiv;
  int ksplit; long sFsp;
};

// ---------------------------------------------------------------------------
// Small-tile direct-global WMMA GEMM (64x64 block, 4 waves, wave = 16x64).
// Used for all hd/A=64 shaped batched GEMMs, with optional split-K.
// ---------------------------------------------------------------------------
template<int WM, int WN, int TM, int TN>
__launch_bounds__(WM * WN * 32)
__global__ void gemm_bf16_wmma(GemmArgs p) {
  const int lane = threadIdx.x & 31;
  const int wave = threadIdx.x >> 5;
  const int wm = wave / WN, wn = wave % WN;
  const int zd = (p.zdiv > 0) ? p.zdiv : 1;
  const int ks = (p.ksplit > 0) ? p.ksplit : 1;
  const int batch = blockIdx.z / ks;
  const int sk = blockIdx.z % ks;
  const long zhi = batch / zd, zlo = batch % zd;
  const int kchunk = p.K / ks;
  const int k0 = sk * kchunk;

  const bf16* A = p.A + zhi * p.sAhi + zlo * p.sAlo;
  const bf16* W = p.W + zhi * p.sWhi + zlo * p.sWlo;

  const int r  = lane & 15;   // row (A) / col (B) within 16-tile
  const int hf = lane >> 4;   // K-half selector per ISA 16-bit layout

  const long rm0 = (long)blockIdx.x * (WM * TM * 16) + wm * (TM * 16);
  const long cn0 = (long)blockIdx.y * (WN * TN * 16) + wn * (TN * 16);

  union Frag { v16bf v; v4u u[2]; };
  v8f acc[TM][TN] = {};

  const bf16* Ab = A + (rm0 + r) * (long)p.lda + hf * 8;
  const bf16* Wb = W + (cn0 + r) * (long)p.ldw + hf * 8;

  for (int kb = k0; kb < k0 + kchunk; kb += 32) {
    Frag a[TM], b[TN];
#pragma unroll
    for (int i = 0; i < TM; ++i) {
      const bf16* q = Ab + (long)i * 16 * p.lda + kb;
      a[i].u[0] = *(const v4u*)q;
      a[i].u[1] = *(const v4u*)(q + 16);
    }
#pragma unroll
    for (int j = 0; j < TN; ++j) {
      const bf16* q = Wb + (long)j * 16 * p.ldw + kb;
      b[j].u[0] = *(const v4u*)q;
      b[j].u[1] = *(const v4u*)(q + 16);
    }
    if (kb + 32 < k0 + kchunk) {
      __builtin_prefetch(Ab + kb + 32, 0, 3);
      __builtin_prefetch(Wb + kb + 32, 0, 3);
    }
#pragma unroll
    for (int i = 0; i < TM; ++i)
#pragma unroll
      for (int j = 0; j < TN; ++j)
        acc[i][j] = __builtin_amdgcn_wmma_f32_16x16x32_bf16(
            false, a[i].v, false, b[j].v, (short)0, acc[i][j], false, false);
  }

  float* OF = p.outF ? (p.outF + zhi * p.sFhi + zlo * p.sFlo + (long)sk * p.sFsp) : (float*)0;
  bf16*  OH = p.outH ? (p.outH + zhi * p.sHhi + zlo * p.sHlo) : (bf16*)0;

#pragma unroll
  for (int i = 0; i < TM; ++i)
#pragma unroll
    for (int j = 0; j < TN; ++j)
#pragma unroll
      for (int e = 0; e < 8; ++e) {
        long row = rm0 + i * 16 + e + hf * 8;
        long col = cn0 + j * 16 + (lane & 15);
        float v = acc[i][j][e] * p.alpha;
        if (p.bias) v += p.bias[col];
        if (OF) OF[row * (long)p.ldf + col] = v;
        if (OH) {
          if (p.transH) OH[col * (long)p.ldh + row] = f2bf(v);
          else          OH[row * (long)p.ldh + col] = f2bf(v);
        }
      }
}

// ---------------------------------------------------------------------------
// Big-tile WMMA GEMM with double-buffered async global->LDS staging.
// Block tile 128x128, 256 threads (8 waves as 2x4), wave tile 64x32.
// Each K-step: 128x32 A-slice + 128x32 B-slice DMA'd to LDS via
// GLOBAL_LOAD_ASYNC_TO_LDS_B128 (4 ops/thread), fenced with s_wait_asynccnt,
// fragments re-read conflict-free (80B padded rows) via ds_load_b128.
// ---------------------------------------------------------------------------
#define LDS_ROW 40              // 32 payload elements + 8 pad (80 bytes)
#define LDS_BUF (128 * LDS_ROW) // elements per buffer

__launch_bounds__(256)
__global__ void gemm_big_async(GemmArgs p) {
  __shared__ __align__(16) bf16 ldsA[2][LDS_BUF];
  __shared__ __align__(16) bf16 ldsB[2][LDS_BUF];

  const int t    = threadIdx.x;
  const int lane = t & 31;
  const int wave = t >> 5;
  const int wm = wave >> 2;     // 0..1  (rows)
  const int wn = wave & 3;      // 0..3  (cols)
  const int zd = (p.zdiv > 0) ? p.zdiv : 1;
  const long zhi = blockIdx.z / zd, zlo = blockIdx.z % zd;

  const bf16* A = p.A + zhi * p.sAhi + zlo * p.sAlo + (long)blockIdx.x * 128 * p.lda;
  const bf16* W = p.W + zhi * p.sWhi + zlo * p.sWlo + (long)blockIdx.y * 128 * p.ldw;

  const unsigned baseA = lds_addr_of(&ldsA[0][0]);
  const unsigned baseB = lds_addr_of(&ldsB[0][0]);

  const int r  = lane & 15;
  const int hf = lane >> 4;

  // per-thread DMA slots: 512 b128 segments per tile, 2 per thread per tile
  const int row0 = t >> 1;                  // 0..127
  const int seg0 = (t & 1) * 2;             // segments {0,1} or {2,3}

  auto issue = [&](int buf, int kb) {
    async_b128(baseA + (unsigned)buf * (LDS_BUF * 2) + row0 * 80 + seg0 * 16,
               A + (long)row0 * p.lda + kb + seg0 * 8);
    async_b128(baseA + (unsigned)buf * (LDS_BUF * 2) + row0 * 80 + (seg0 + 1) * 16,
               A + (long)row0 * p.lda + kb + (seg0 + 1) * 8);
    async_b128(baseB + (unsigned)buf * (LDS_BUF * 2) + row0 * 80 + seg0 * 16,
               W + (long)row0 * p.ldw + kb + seg0 * 8);
    async_b128(baseB + (unsigned)buf * (LDS_BUF * 2) + row0 * 80 + (seg0 + 1) * 16,
               W + (long)row0 * p.ldw + kb + (seg0 + 1) * 8);
  };

  union Frag { v16bf v; v4u u[2]; };
  v8f acc[4][2] = {};

  issue(0, 0);                               // prologue: stage first K-slice

  int it = 0;
  for (int kb = 0; kb < p.K; kb += 32, it ^= 1) {
    const bool has_next = (kb + 32) < p.K;
    __syncthreads();                         // buf[it^1] free of readers
    if (has_next) issue(it ^ 1, kb + 32);    // overlap DMA with compute
    if (has_next) asm volatile("s_wait_asynccnt 0x4" ::: "memory");
    else          asm volatile("s_wait_asynccnt 0x0" ::: "memory");
    __syncthreads();                         // all waves' buf[it] landed

    Frag a[4], b[2];
#pragma unroll
    for (int i = 0; i < 4; ++i) {
      const bf16* la = &ldsA[it][(wm * 64 + i * 16 + r) * LDS_ROW + hf * 8];
      a[i].u[0] = *(const v4u*)la;
      a[i].u[1] = *(const v4u*)(la + 16);
    }
#pragma unroll
    for (int j = 0; j < 2; ++j) {
      const bf16* lb = &ldsB[it][(wn * 32 + j * 16 + r) * LDS_ROW + hf * 8];
      b[j].u[0] = *(const v4u*)lb;
      b[j].u[1] = *(const v4u*)(lb + 16);
    }
#pragma unroll
    for (int i = 0; i < 4; ++i)
#pragma unroll
      for (int j = 0; j < 2; ++j)
        acc[i][j] = __builtin_amdgcn_wmma_f32_16x16x32_bf16(
            false, a[i].v, false, b[j].v, (short)0, acc[i][j], false, false);
  }

  const long rm0 = (long)blockIdx.x * 128 + wm * 64;
  const long cn0 = (long)blockIdx.y * 128 + wn * 32;
  float* OF = p.outF ? (p.outF + zhi * p.sFhi + zlo * p.sFlo) : (float*)0;
  bf16*  OH = p.outH ? (p.outH + zhi * p.sHhi + zlo * p.sHlo) : (bf16*)0;

#pragma unroll
  for (int i = 0; i < 4; ++i)
#pragma unroll
    for (int j = 0; j < 2; ++j)
#pragma unroll
      for (int e = 0; e < 8; ++e) {
        long row = rm0 + i * 16 + e + hf * 8;
        long col = cn0 + j * 16 + r;
        float v = acc[i][j][e] * p.alpha;
        if (p.bias) v += p.bias[col];
        if (OF) OF[row * (long)p.ldf + col] = v;
        if (OH) OH[row * (long)p.ldh + col] = f2bf(v);
      }
}

// ---------------------------------------------------------------------------
// Softmax over long rows (len = T = 4096), one 256-thread block per row.
// ---------------------------------------------------------------------------
__global__ void softmax_rows(const float* __restrict__ in, bf16* __restrict__ outb, int len) {
  const long row = blockIdx.x;
  const float* p = in + row * (long)len;
  bf16* o = outb + row * (long)len;
  __shared__ float red[256];
  const int tid = threadIdx.x;

  float mx = -3.0e38f;
  for (int i = tid; i < len; i += 256) mx = fmaxf(mx, p[i]);
  red[tid] = mx; __syncthreads();
  for (int s = 128; s > 0; s >>= 1) {
    if (tid < s) red[tid] = fmaxf(red[tid], red[tid + s]);
    __syncthreads();
  }
  mx = red[0]; __syncthreads();

  float sum = 0.f;
  for (int i = tid; i < len; i += 256) sum += __expf(p[i] - mx);
  red[tid] = sum; __syncthreads();
  for (int s = 128; s > 0; s >>= 1) {
    if (tid < s) red[tid] += red[tid + s];
    __syncthreads();
  }
  const float inv = 1.f / red[0];
  for (int i = tid; i < len; i += 256) o[i] = f2bf(__expf(p[i] - mx) * inv);
}

// ---------------------------------------------------------------------------
// Softmax over rows of length 64 (wave32 per row, 2 values per lane).
// ---------------------------------------------------------------------------
__global__ void softmax64(const float* __restrict__ in, float* __restrict__ outF,
                          bf16* __restrict__ outb) {
  const int lane = threadIdx.x & 31;
  const long row = (long)blockIdx.x * 8 + (threadIdx.x >> 5);
  const float2 v = ((const float2*)(in + row * 64))[lane];
  float mx = fmaxf(v.x, v.y);
#pragma unroll
  for (int m = 16; m >= 1; m >>= 1) mx = fmaxf(mx, __shfl_xor(mx, m, 32));
  float ex = __expf(v.x - mx), ey = __expf(v.y - mx);
  float s = ex + ey;
#pragma unroll
  for (int m = 16; m >= 1; m >>= 1) s += __shfl_xor(s, m, 32);
  const float inv = 1.f / s;
  ex *= inv; ey *= inv;
  ((float2*)(outF + row * 64))[lane] = make_float2(ex, ey);
  unsigned u = (unsigned)__builtin_bit_cast(unsigned short, f2bf(ex))
             | ((unsigned)__builtin_bit_cast(unsigned short, f2bf(ey)) << 16);
  ((unsigned*)(outb + row * 64))[lane] = u;
}

// ---------------------------------------------------------------------------
// Reduce 8 split-K partials of c1 (f32 [z][8][64*64]) and pack into bf16
// row-major + transposed copies. Fixed summation order => deterministic.
// ---------------------------------------------------------------------------
__global__ void c1pack(const float* __restrict__ c1part, bf16* __restrict__ c1b,
                       bf16* __restrict__ c1Tb) {
  const int z = blockIdx.x;
  const float* src = c1part + (long)z * 8 * 4096;
  for (int e = threadIdx.x; e < 4096; e += blockDim.x) {
    float v = 0.f;
#pragma unroll
    for (int s = 0; s < 8; ++s) v += src[s * 4096 + e];
    int a = e >> 6, d = e & 63;
    bf16 h = f2bf(v);
    c1b [(long)z * 4096 + e]          = h;
    c1Tb[(long)z * 4096 + d * 64 + a] = h;
  }
}

// ---------------------------------------------------------------------------
// Host orchestration
// ---------------------------------------------------------------------------
extern "C" void kernel_launch(void* const* d_in, const int* in_sizes, int n_in,
                              void* d_out, int out_size, void* d_ws, size_t ws_size,
                              hipStream_t stream) {
  (void)in_sizes; (void)n_in; (void)out_size; (void)ws_size;

  enum { Bn = 4, T = 4096, C = 1024, NH = 16, NA = 64, HD = 64 };
  const long BT = (long)Bn * T;            // 16384 rows
  const long NX = BT * C;                  // 16,777,216
  const float scale = 0.125f;              // 1/sqrt(64)

  const float* x      = (const float*)d_in[0];
  const float* Wqkv   = (const float*)d_in[1];
  const float* bqkv   = (const float*)d_in[2];
  const float* Wq     = (const float*)d_in[3];
  const float* bq     = (const float*)d_in[4];
  const float* Wk     = (const float*)d_in[5];
  const float* bk     = (const float*)d_in[6];
  const float* Wv     = (const float*)d_in[7];
  const float* bv     = (const float*)d_in[8];
  const float* agents = (const float*)d_in[9];
  const float* Wout   = (const float*)d_in[10];
  const float* bout   = (const float*)d_in[11];
  const float* Wproj  = (const float*)d_in[12];
  const float* bproj  = (const float*)d_in[13];

  size_t off = 0;
  char* base = (char*)d_ws;
  auto alloc = [&](size_t bytes) -> void* {
    void* p = base + off;
    off += (bytes + 255) & ~(size_t)255;
    return p;
  };

  bf16* xb     = (bf16*)alloc(NX * 2);
  bf16* wqkvb  = (bf16*)alloc((size_t)3 * C * C * 2);
  bf16* wqb    = (bf16*)alloc((size_t)HD * HD * 2);
  bf16* wkb    = (bf16*)alloc((size_t)HD * HD * 2);
  bf16* wvb    = (bf16*)alloc((size_t)HD * HD * 2);
  bf16* woutb  = (bf16*)alloc((size_t)C * C * 2);
  bf16* wprojb = (bf16*)alloc((size_t)C * C * 2);
  bf16* agb    = (bf16*)alloc((size_t)NH * NA * HD * 2);
  bf16* qkvb   = (bf16*)alloc((size_t)BT * 3 * C * 2);
  bf16* qab    = (bf16*)alloc((size_t)NH * NA * HD * 2);
  bf16* kab    = (bf16*)alloc((size_t)NH * BT * HD * 2);          // [h][b*T+t][d]
  bf16* vaTb   = (bf16*)alloc((size_t)NH * HD * BT * 2);          // [h][d][b*T+t]
  float* s1f   = (float*)alloc((size_t)Bn * NH * NA * T * 4);     // [z][a][t]
  bf16* p1b    = (bf16*)alloc((size_t)Bn * NH * NA * T * 2);
  float* c1part= (float*)alloc((size_t)Bn * NH * 8 * NA * HD * 4);// [z][sk][a][d]
  bf16* c1b    = (bf16*)alloc((size_t)Bn * NH * NA * HD * 2);
  bf16* c1Tb   = (bf16*)alloc((size_t)Bn * NH * NA * HD * 2);
  float* s2f   = (float*)alloc((size_t)Bn * NH * T * NA * 4);     // [z][t][a]
  bf16* p2b    = (bf16*)alloc((size_t)Bn * NH * T * NA * 2);
  bf16* c2b    = (bf16*)alloc((size_t)BT * C * 2);                // [b*T+t][h*64+d]
  bf16* y1b    = (bf16*)alloc((size_t)BT * C * 2);

  auto cvt = [&](const float* src, bf16* dst, long n) {
    cvt_f32_bf16<<<dim3(2048), dim3(256), 0, stream>>>(src, dst, n);
  };
  cvt(x, xb, NX);
  cvt(Wqkv, wqkvb, 3L * C * C);
  cvt(Wq, wqb, (long)HD * HD);
  cvt(Wk, wkb, (long)HD * HD);
  cvt(Wv, wvb, (long)HD * HD);
  cvt(Wout, woutb, (long)C * C);
  cvt(Wproj, wprojb, (long)C * C);
  cvt(agents, agb, (long)NH * NA * HD);

  // ---- 1) qkv = x @ Wqkv^T + bqkv  (async LDS big tile) ----
  {
    GemmArgs g = {};
    g.A = xb; g.lda = C;
    g.W = wqkvb; g.ldw = C;
    g.bias = bqkv;
    g.outH = qkvb; g.ldh = 3 * C;
    g.K = C; g.alpha = 1.f; g.zdiv = NH; g.ksplit = 1;
    gemm_big_async<<<dim3(BT / 128, 3 * C / 128, 1), 256, 0, stream>>>(g);
  }

  // ---- 2) q_a = agents @ Wq^T + bq ----
  {
    GemmArgs g = {};
    g.A = agb; g.lda = HD;
    g.W = wqb; g.ldw = HD;
    g.bias = bq;
    g.outH = qab; g.ldh = HD;
    g.K = HD; g.alpha = 1.f; g.zdiv = NH; g.ksplit = 1;
    gemm_bf16_wmma<4, 1, 1, 4><<<dim3(NH * NA / 64, 1, 1), 128, 0, stream>>>(g);
  }

  // ---- 3) k_a = k @ Wk^T + bk  (batched over heads) ----
  {
    GemmArgs g = {};
    g.A = qkvb + C; g.sAlo = HD; g.lda = 3 * C;
    g.W = wkb; g.ldw = HD;
    g.bias = bk;
    g.outH = kab; g.sHlo = BT * HD; g.ldh = HD;
    g.K = HD; g.alpha = 1.f; g.zdiv = NH; g.ksplit = 1;
    gemm_bf16_wmma<4, 1, 1, 4><<<dim3(BT / 64, 1, NH), 128, 0, stream>>>(g);
  }

  // ---- 4) v_a = v @ Wv^T + bv, stored transposed: vaT[h][d][b*T+t] ----
  {
    GemmArgs g = {};
    g.A = qkvb + 2 * C; g.sAlo = HD; g.lda = 3 * C;
    g.W = wvb; g.ldw = HD;
    g.bias = bv;
    g.outH = vaTb; g.sHlo = (long)HD * BT; g.ldh = (int)BT; g.transH = 1;
    g.K = HD; g.alpha = 1.f; g.zdiv = NH; g.ksplit = 1;
    gemm_bf16_wmma<4, 1, 1, 4><<<dim3(BT / 64, 1, NH), 128, 0, stream>>>(g);
  }

  // ---- 5) s1 raw = scale * q_a @ k_a^T   (z=(b,h); [64 x T]) ----
  {
    GemmArgs g = {};
    g.A = qab; g.sAlo = (long)NA * HD; g.lda = HD;
    g.W = kab; g.sWhi = (long)T * HD; g.sWlo = (long)BT * HD; g.ldw = HD;
    g.alpha = scale;
    g.outF = s1f; g.sFhi = (long)NH * NA * T; g.sFlo = (long)NA * T; g.ldf = T;
    g.K = HD; g.zdiv = NH; g.ksplit = 1;
    gemm_bf16_wmma<4, 1, 1, 4><<<dim3(1, T / 64, Bn * NH), 128, 0, stream>>>(g);
  }

  // ---- 6) softmax over T, emit bf16 P1 ----
  softmax_rows<<<dim3(Bn * NH * NA), 256, 0, stream>>>(s1f, p1b, T);

  // ---- 7) c1 = P1 @ v_a   (z=(b,h); [64 x 64], K=T, split-K x8) ----
  {
    GemmArgs g = {};
    g.A = p1b; g.sAhi = (long)NH * NA * T; g.sAlo = (long)NA * T; g.lda = T;
    g.W = vaTb; g.sWhi = (long)T; g.sWlo = (long)HD * BT; g.ldw = (int)BT;
    g.alpha = 1.f;
    g.outF = c1part;
    g.sFhi = (long)NH * 8 * NA * HD; g.sFlo = (long)8 * NA * HD; g.ldf = HD;
    g.sFsp = (long)NA * HD;
    g.K = T; g.zdiv = NH; g.ksplit = 8;
    gemm_bf16_wmma<4, 1, 1, 4><<<dim3(1, 1, Bn * NH * 8), 128, 0, stream>>>(g);
  }

  // ---- 8) reduce split-K partials, pack c1 -> bf16 (+ transposed) ----
  c1pack<<<dim3(Bn * NH), 256, 0, stream>>>(c1part, c1b, c1Tb);

  // ---- 9) s2 raw = scale * q @ c1^T   (z=(b,h); [T x 64]) ----
  {
    GemmArgs g = {};
    g.A = qkvb; g.sAhi = (long)T * 3 * C; g.sAlo = HD; g.lda = 3 * C;
    g.W = c1b; g.sWhi = (long)NH * NA * HD; g.sWlo = (long)NA * HD; g.ldw = HD;
    g.alpha = scale;
    g.outF = s2f; g.sFhi = (long)NH * T * NA; g.sFlo = (long)T * NA; g.ldf = NA;
    g.K = HD; g.zdiv = NH; g.ksplit = 1;
    gemm_bf16_wmma<4, 1, 1, 4><<<dim3(T / 64, 1, Bn * NH), 128, 0, stream>>>(g);
  }

  // ---- 10) softmax over A=64: f32 s2 output + bf16 P2 ----
  float* s2out = (float*)d_out + NX;   // outputs: y then s2
  softmax64<<<dim3((unsigned)(Bn * NH * T / 8)), 256, 0, stream>>>(s2f, s2out, p2b);

  // ---- 11) c2 = P2 @ c1  -> bf16 [b*T+t][h*64+d] ----
  {
    GemmArgs g = {};
    g.A = p2b; g.sAhi = (long)NH * T * NA; g.sAlo = (long)T * NA; g.lda = NA;
    g.W = c1Tb; g.sWhi = (long)NH * HD * NA; g.sWlo = (long)HD * NA; g.ldw = NA;
    g.alpha = 1.f;
    g.outH = c2b; g.sHhi = (long)T * C; g.sHlo = HD; g.ldh = C;
    g.K = NA; g.zdiv = NH; g.ksplit = 1;
    gemm_bf16_wmma<4, 1, 1, 4><<<dim3(T / 64, 1, Bn * NH), 128, 0, stream>>>(g);
  }

  // ---- 12) y1 = c2 @ Wout^T + bout  (async LDS big tile) ----
  {
    GemmArgs g = {};
    g.A = c2b; g.lda = C;
    g.W = woutb; g.ldw = C;
    g.bias = bout;
    g.outH = y1b; g.ldh = C;
    g.K = C; g.alpha = 1.f; g.zdiv = NH; g.ksplit = 1;
    gemm_big_async<<<dim3(BT / 128, C / 128, 1), 256, 0, stream>>>(g);
  }

  // ---- 13) y = y1 @ Wproj^T + bproj -> f32 d_out  (async LDS big tile) ----
  {
    GemmArgs g = {};
    g.A = y1b; g.lda = C;
    g.W = wprojb; g.ldw = C;
    g.bias = bproj;
    g.outF = (float*)d_out; g.ldf = C;
    g.K = C; g.alpha = 1.f; g.zdiv = NH; g.ksplit = 1;
    gemm_big_async<<<dim3(BT / 128, C / 128, 1), 256, 0, stream>>>(g);
  }
}
